// HyperNetwork_15522011808526
// MI455X (gfx1250) — matrix operand order repaired and checked by reading, not compiled, not run on hardware
//
#include <hip/hip_runtime.h>
#include <hip/hip_bf16.h>
#include <math.h>

typedef __attribute__((ext_vector_type(16))) _Float16 v16h;
typedef __attribute__((ext_vector_type(8)))  _Float16 v8h;
typedef __attribute__((ext_vector_type(8)))  float    v8f;

#define NUM_TREES   15
#define NUM_INNER   15
#define NUM_LEAVES  16
#define INPUT_DIM   256
#define NUM_CLASSES 100
#define PARAMS_PER_TREE 5455   // 15*(256+1) + 16*100
#define TOTAL_PARAMS (NUM_TREES * PARAMS_PER_TREE)  // 81825
#define BATCH       131072

// ---- workspace layout (bytes) ----
#define WS_PARTIAL 0          // 256*256 f32 partial column sums
#define WS_H       262144     // 128 f32 hidden vector
#define WS_PARAMS  263680     // 81825 f32 generated params
#define WS_BIAS    591104     // 240 f32 inner biases (15 trees x 16, pad)
#define WS_W16     592128     // 15*16*256 f16 inner weights (node 15 = 0)
#define WS_LEAFP   715008     // 112*256 f16: [class][tree*16+leaf], w_t folded in

// LDS row pitch (f32) for the 32-row decision/path buffer
#define ROWF 264

// ---------------- K1: partial column sums of context ----------------
__global__ __launch_bounds__(256) void k1_colsum(const float* __restrict__ ctxin,
                                                 float* __restrict__ partial) {
  const int t = threadIdx.x, b = blockIdx.x;
  const float* p = ctxin + (size_t)b * 512 * INPUT_DIM + t;
  float s = 0.f;
  for (int r = 0; r < 512; ++r) s += p[(size_t)r * INPUT_DIM];
  partial[b * 256 + t] = s;
}

// ---------------- K2: finish mean + 4-layer MLP (single block) ----------------
__global__ __launch_bounds__(256) void k2_mlp(const float* __restrict__ partial,
    const float* __restrict__ cw1, const float* __restrict__ cb1,
    const float* __restrict__ cw2, const float* __restrict__ cb2,
    const float* __restrict__ hw1, const float* __restrict__ hb1,
    const float* __restrict__ hw2, const float* __restrict__ hb2,
    float* __restrict__ hout) {
  __shared__ float ctx[256], h1[32], h2[64], h3[128];
  const int t = threadIdx.x;
  float s = 0.f;
  for (int i = 0; i < 256; ++i) s += partial[i * 256 + t];
  ctx[t] = s * (1.0f / (float)BATCH);
  __syncthreads();
  if (t < 32)  { float a = cb1[t]; const float* w = cw1 + t * 256; for (int k = 0; k < 256; ++k) a += w[k] * ctx[k]; h1[t] = fmaxf(a, 0.f); }
  __syncthreads();
  if (t < 64)  { float a = cb2[t]; const float* w = cw2 + t * 32;  for (int k = 0; k < 32;  ++k) a += w[k] * h1[k];  h2[t] = fmaxf(a, 0.f); }
  __syncthreads();
  if (t < 128) { float a = hb1[t]; const float* w = hw1 + t * 64;  for (int k = 0; k < 64;  ++k) a += w[k] * h2[k];  h3[t] = fmaxf(a, 0.f); }
  __syncthreads();
  if (t < 128) { float a = hb2[t]; const float* w = hw2 + t * 128; for (int k = 0; k < 128; ++k) a += w[k] * h3[k];  hout[t] = fmaxf(a, 0.f); }
}

// ---------------- K3: params = hn_w3 @ h + b3  (42MB, fully coalesced) ----------------
// Each wave sweeps 16 consecutive rows; per row, lane l reads cols 4l..4l+3
// (one contiguous 512B transaction), then a 5-step shfl_xor reduction.
__global__ __launch_bounds__(256) void k3_params(const float* __restrict__ w3,
    const float* __restrict__ b3, const float* __restrict__ h,
    float* __restrict__ params) {
  __shared__ float hs[128];
  const int t = threadIdx.x;
  if (t < 128) hs[t] = h[t];
  __syncthreads();
  const int lane = t & 31, wave = t >> 5;
  const int rowbase = (blockIdx.x * 8 + wave) * 16;
  const float hx = hs[lane * 4 + 0], hy = hs[lane * 4 + 1];
  const float hz = hs[lane * 4 + 2], hw = hs[lane * 4 + 3];
  #pragma unroll 4
  for (int k = 0; k < 16; ++k) {
    const int r = rowbase + k;
    if (r >= TOTAL_PARAMS) break;
    const float4 v = *((const float4*)(w3 + (size_t)r * 128) + lane);
    float a = v.x * hx + v.y * hy + v.z * hz + v.w * hw;
    a += __shfl_xor(a, 1, 32);
    a += __shfl_xor(a, 2, 32);
    a += __shfl_xor(a, 4, 32);
    a += __shfl_xor(a, 8, 32);
    a += __shfl_xor(a, 16, 32);
    if (lane == 0) params[r] = a + b3[r];
  }
}

// ---------------- K4a: pack inner weights to f16 B layout + biases ----------------
__global__ __launch_bounds__(256) void k4a_pack_w(const float* __restrict__ params,
    _Float16* __restrict__ W16, float* __restrict__ biasb) {
  const int i = blockIdx.x * 256 + threadIdx.x;
  if (i < NUM_TREES * 16 * 256) {
    const int t = i >> 12, rem = i & 4095, n = rem >> 8, k = rem & 255;
    float v = (n < NUM_INNER) ? params[t * PARAMS_PER_TREE + n * 256 + k] : 0.f;
    W16[i] = (_Float16)v;
  }
  if (i < 240) {
    const int t = i >> 4, n = i & 15;
    biasb[i] = (n < NUM_INNER) ? params[t * PARAMS_PER_TREE + 3840 + n] : 0.f;
  }
}

// ---------------- K4b: leaf softmax * softmax(tree_w) -> leafP[class][treeleaf] f16 ----------------
__global__ __launch_bounds__(128) void k4b_leaf(const float* __restrict__ params,
    const float* __restrict__ tw, _Float16* __restrict__ leafP) {
  __shared__ float red[128];
  const int t = blockIdx.x >> 4, leaf = blockIdx.x & 15, c = threadIdx.x;
  const int k = blockIdx.x;  // column index = tree*16 + leaf (240..255 = pad)
  if (t >= NUM_TREES) { if (c < 112) leafP[c * 256 + k] = (_Float16)0.f; return; }
  float wmx = -1e30f;
  for (int j = 0; j < NUM_TREES; ++j) wmx = fmaxf(wmx, tw[j]);
  float wsum = 0.f;
  for (int j = 0; j < NUM_TREES; ++j) wsum += __expf(tw[j] - wmx);
  const float wt = __expf(tw[t] - wmx) / wsum;
  const float* row = params + t * PARAMS_PER_TREE + 3855 + leaf * NUM_CLASSES;
  const float v = (c < NUM_CLASSES) ? row[c] : -1e30f;
  red[c] = v; __syncthreads();
  for (int s = 64; s > 0; s >>= 1) { if (c < s) red[c] = fmaxf(red[c], red[c + s]); __syncthreads(); }
  const float mx = red[0]; __syncthreads();
  const float e = (c < NUM_CLASSES) ? __expf(v - mx) : 0.f;
  red[c] = e; __syncthreads();
  for (int s = 64; s > 0; s >>= 1) { if (c < s) red[c] += red[c + s]; __syncthreads(); }
  const float scale = wt / red[0];
  if (c < 112) leafP[c * 256 + k] = (_Float16)((c < NUM_CLASSES) ? e * scale : 0.f);
}

// fragment packers
__device__ inline v16h pack16(float4 g0, float4 g1, float4 g2, float4 g3) {
  v16h a;
  a[0]=(_Float16)g0.x; a[1]=(_Float16)g0.y; a[2]=(_Float16)g0.z; a[3]=(_Float16)g0.w;
  a[4]=(_Float16)g1.x; a[5]=(_Float16)g1.y; a[6]=(_Float16)g1.z; a[7]=(_Float16)g1.w;
  a[8]=(_Float16)g2.x; a[9]=(_Float16)g2.y; a[10]=(_Float16)g2.z; a[11]=(_Float16)g2.w;
  a[12]=(_Float16)g3.x; a[13]=(_Float16)g3.y; a[14]=(_Float16)g3.z; a[15]=(_Float16)g3.w;
  return a;
}
__device__ inline v16h packb(v8h b0, v8h b1) {
  v16h b;
  #pragma unroll
  for (int e = 0; e < 8; ++e) { b[e] = b0[e]; b[8 + e] = b1[e]; }
  return b;
}

// ---------------- K5: 32 samples per wave: decisions GEMM -> sigmoid -> paths -> leaf GEMM ----------------
// Two 16-row sub-tiles share every B fragment (2 WMMAs per B load) to halve L2 traffic.
__global__ __launch_bounds__(128) void k5_forest(const float* __restrict__ x,
    const _Float16* __restrict__ W16, const float* __restrict__ biasb,
    const _Float16* __restrict__ leafP, float* __restrict__ out) {
  extern __shared__ float lds[];
  const int lane = threadIdx.x & 31;
  const int wave = threadIdx.x >> 5;
  const int tile = blockIdx.x * 4 + wave;          // 32 samples per wave-tile
  float* buf = lds + wave * (32 * ROWF);
  const int m16 = lane & 15;
  const int half = lane >> 4;

  // ---- GEMM1 A fragments: 2 sub-tiles x 8 k-chunks (f32 -> f16, ISA A layout)
  v16h afrag[2][8];
  #pragma unroll
  for (int sub = 0; sub < 2; ++sub) {
    const float* xrow = x + (size_t)(tile * 32 + sub * 16 + m16) * INPUT_DIM;
    #pragma unroll
    for (int kc = 0; kc < 8; ++kc) {
      const int base = kc * 32 + half * 8;
      afrag[sub][kc] = pack16(*(const float4*)(xrow + base),      *(const float4*)(xrow + base + 4),
                              *(const float4*)(xrow + base + 16), *(const float4*)(xrow + base + 20));
    }
  }

  // ---- decisions: per tree, 8 B-frags x 2 wmma, + bias, sigmoid -> LDS
  for (int t = 0; t < NUM_TREES; ++t) {
    v8f acc0 = {}, acc1 = {};
    const _Float16* wrow = W16 + (size_t)(t * 16 + m16) * 256;
    #pragma unroll
    for (int kc = 0; kc < 8; ++kc) {
      const int kb = kc * 32 + half * 8;
      v16h b = packb(*(const v8h*)(wrow + kb), *(const v8h*)(wrow + kb + 16));
      acc0 = __builtin_amdgcn_wmma_f32_16x16x32_f16(false, afrag[0][kc], false, b,
                                                    (short)0, acc0, false, false);
      acc1 = __builtin_amdgcn_wmma_f32_16x16x32_f16(false, afrag[1][kc], false, b,
                                                    (short)0, acc1, false, false);
    }
    const float bias = biasb[t * 16 + m16];
    #pragma unroll
    for (int r = 0; r < 8; ++r) {
      const float v0 = acc0[r] + bias, v1 = acc1[r] + bias;
      buf[(r + 8 * half) * ROWF + t * 16 + m16]      = 1.0f / (1.0f + __expf(-v0));
      buf[(16 + r + 8 * half) * ROWF + t * 16 + m16] = 1.0f / (1.0f + __expf(-v1));
    }
  }
  __syncthreads();

  // ---- soft tree traversal: lane owns one of the 32 samples, walks all 15 trees
  {
    const int m = lane;
    for (int t = 0; t < NUM_TREES; ++t) {
      float s[NUM_INNER];
      #pragma unroll
      for (int n = 0; n < NUM_INNER; ++n) s[n] = buf[m * ROWF + t * 16 + n];
      float pv[NUM_LEAVES];
      #pragma unroll
      for (int l = 0; l < NUM_LEAVES; ++l) {
        float p = 1.0f; int node = 0;
        #pragma unroll
        for (int d = 0; d < 4; ++d) {
          const int bit = (l >> (3 - d)) & 1;
          p *= bit ? (1.0f - s[node]) : s[node];
          node = 2 * node + 1 + bit;
        }
        pv[l] = p;
      }
      #pragma unroll
      for (int l = 0; l < NUM_LEAVES; ++l) buf[m * ROWF + t * 16 + l] = pv[l];
    }
    #pragma unroll
    for (int k = 240; k < 256; ++k) buf[m * ROWF + k] = 0.0f;  // zero K-pad
  }
  __syncthreads();

  // ---- GEMM2: path_probs(32x256) @ leafP(256x112), trees folded into K
  v16h pfrag[2][8];
  #pragma unroll
  for (int sub = 0; sub < 2; ++sub) {
    const float* prow = buf + (sub * 16 + m16) * ROWF;
    #pragma unroll
    for (int kc = 0; kc < 8; ++kc) {
      const int base = kc * 32 + half * 8;
      pfrag[sub][kc] = pack16(*(const float4*)(prow + base),      *(const float4*)(prow + base + 4),
                              *(const float4*)(prow + base + 16), *(const float4*)(prow + base + 20));
    }
  }
  for (int ct = 0; ct < 7; ++ct) {
    v8f acc0 = {}, acc1 = {};
    const _Float16* lrow = leafP + (size_t)(ct * 16 + m16) * 256;
    #pragma unroll
    for (int kc = 0; kc < 8; ++kc) {
      const int kb = kc * 32 + half * 8;
      v16h b = packb(*(const v8h*)(lrow + kb), *(const v8h*)(lrow + kb + 16));
      acc0 = __builtin_amdgcn_wmma_f32_16x16x32_f16(false, pfrag[0][kc], false, b,
                                                    (short)0, acc0, false, false);
      acc1 = __builtin_amdgcn_wmma_f32_16x16x32_f16(false, pfrag[1][kc], false, b,
                                                    (short)0, acc1, false, false);
    }
    const int c = ct * 16 + m16;
    if (c < NUM_CLASSES) {
      #pragma unroll
      for (int r = 0; r < 8; ++r) {
        out[(size_t)(tile * 32 + r + 8 * half) * NUM_CLASSES + c]      = acc0[r];
        out[(size_t)(tile * 32 + 16 + r + 8 * half) * NUM_CLASSES + c] = acc1[r];
      }
    }
  }
}

extern "C" void kernel_launch(void* const* d_in, const int* in_sizes, int n_in,
                              void* d_out, int out_size, void* d_ws, size_t ws_size,
                              hipStream_t stream) {
  const float* x        = (const float*)d_in[0];
  const float* context  = (const float*)d_in[1];
  const float* ce_w1    = (const float*)d_in[2];
  const float* ce_b1    = (const float*)d_in[3];
  const float* ce_w2    = (const float*)d_in[4];
  const float* ce_b2    = (const float*)d_in[5];
  const float* hn_w1    = (const float*)d_in[6];
  const float* hn_b1    = (const float*)d_in[7];
  const float* hn_w2    = (const float*)d_in[8];
  const float* hn_b2    = (const float*)d_in[9];
  const float* hn_w3    = (const float*)d_in[10];
  const float* hn_b3    = (const float*)d_in[11];
  const float* tree_w   = (const float*)d_in[12];

  char* ws = (char*)d_ws;
  float*    partial = (float*)(ws + WS_PARTIAL);
  float*    hvec    = (float*)(ws + WS_H);
  float*    params  = (float*)(ws + WS_PARAMS);
  float*    biasb   = (float*)(ws + WS_BIAS);
  _Float16* W16     = (_Float16*)(ws + WS_W16);
  _Float16* leafP   = (_Float16*)(ws + WS_LEAFP);
  float* out = (float*)d_out;

  k1_colsum<<<256, 256, 0, stream>>>(context, partial);
  k2_mlp<<<1, 256, 0, stream>>>(partial, ce_w1, ce_b1, ce_w2, ce_b2,
                                hn_w1, hn_b1, hn_w2, hn_b2, hvec);
  // 8 waves/block * 16 rows/wave = 128 rows/block
  k3_params<<<(TOTAL_PARAMS + 127) / 128, 256, 0, stream>>>(hn_w3, hn_b3, hvec, params);
  k4a_pack_w<<<240, 256, 0, stream>>>(params, W16, biasb);
  k4b_leaf<<<256, 128, 0, stream>>>(params, tree_w, leafP);
  // 4 waves/block * 32 samples/wave = 128 samples/block
  k5_forest<<<BATCH / 128, 128, 4 * 32 * ROWF * sizeof(float), stream>>>(
      x, W16, biasb, leafP, out);
}
